// SDTPair_61770219651230
// MI455X (gfx1250) — compile-verified, matrix operand order checked by reading, not run on hardware
//
#include <hip/hip_runtime.h>
#include <math.h>

typedef __bf16 bf16;
typedef __attribute__((ext_vector_type(8)))  bf16  v8bf;
typedef __attribute__((ext_vector_type(16))) bf16  v16bf;
typedef __attribute__((ext_vector_type(8)))  float v8f;
typedef __attribute__((ext_vector_type(4)))  unsigned u32x4;
typedef __attribute__((ext_vector_type(8)))  unsigned u32x8;

#define DMODEL 2048
#define NHEAD  16
#define DHEAD  128
#define FFD    8192
#define PFFD   1024
#define BATCH  2
#define SEQ    2048
#define KSEL   1024
#define EPSV   1e-6f

// ---------------------------------------------------------------------------
// Fragment helper: build a v16bf A/B fragment from two 16-byte LDS/global
// chunks at +0 and +16 elements (matches CDNA5 16-bit WMMA VGPR layout:
// lanes 0-15 hold K {0..7,16..23}, lanes 16-31 hold K {8..15,24..31}).
// ---------------------------------------------------------------------------
__device__ __forceinline__ v16bf ld_frag(const bf16* p) {
  v8bf lo = *(const v8bf*)p;
  v8bf hi = *(const v8bf*)(p + 16);
  v16bf r;
#pragma unroll
  for (int i = 0; i < 8; ++i) { r[i] = lo[i]; r[i + 8] = hi[i]; }
  return r;
}

// LDS byte offset of a __shared__ object (generic-pointer low 32 bits)
__device__ __forceinline__ unsigned lds_off(const void* p) {
  return (unsigned)(unsigned long long)p;
}

// ---------------------------------------------------------------------------
// f32 -> bf16 convert (grid-stride)
// ---------------------------------------------------------------------------
__global__ void k_cvt(const float* __restrict__ s, bf16* __restrict__ d, size_t n) {
  for (size_t i = (size_t)blockIdx.x * blockDim.x + threadIdx.x; i < n;
       i += (size_t)gridDim.x * blockDim.x)
    d[i] = (bf16)s[i];
}

// ---------------------------------------------------------------------------
// RMSNorm: one block per row, bf16 output for WMMA GEMM consumption
// ---------------------------------------------------------------------------
__global__ __launch_bounds__(256) void k_rmsnorm(const float* __restrict__ x,
                                                 const float* __restrict__ w,
                                                 bf16* __restrict__ out, int Dn) {
  __shared__ float red[256];
  const int row = blockIdx.x;
  const size_t base = (size_t)row * Dn;
  float s = 0.0f;
  for (int c = threadIdx.x; c < Dn; c += 256) { float v = x[base + c]; s += v * v; }
  red[threadIdx.x] = s; __syncthreads();
  for (int k = 128; k > 0; k >>= 1) {
    if (threadIdx.x < k) red[threadIdx.x] += red[threadIdx.x + k];
    __syncthreads();
  }
  const float r = rsqrtf(red[0] / (float)Dn + EPSV);
  for (int c = threadIdx.x; c < Dn; c += 256)
    out[base + c] = (bf16)(x[base + c] * r * w[c]);
}

// Residual add + RMSNorm fused: h = x + dlt ; out_bf = rms(h) * w
__global__ __launch_bounds__(256) void k_resid_rms(const float* __restrict__ x,
                                                   const float* __restrict__ dlt,
                                                   const float* __restrict__ w,
                                                   float* __restrict__ h,
                                                   bf16* __restrict__ out, int Dn) {
  __shared__ float red[256];
  const int row = blockIdx.x;
  const size_t base = (size_t)row * Dn;
  float s = 0.0f;
  for (int c = threadIdx.x; c < Dn; c += 256) {
    float v = x[base + c] + dlt[base + c];
    h[base + c] = v;
    s += v * v;
  }
  red[threadIdx.x] = s; __syncthreads();
  for (int k = 128; k > 0; k >>= 1) {
    if (threadIdx.x < k) red[threadIdx.x] += red[threadIdx.x + k];
    __syncthreads();
  }
  const float r = rsqrtf(red[0] / (float)Dn + EPSV);
  for (int c = threadIdx.x; c < Dn; c += 256)
    out[base + c] = (bf16)(h[base + c] * r * w[c]);
}

// ---------------------------------------------------------------------------
// bf16 WMMA GEMM: C[M,N] f32 = A[M,K] bf16 (row major) x B[K,N] bf16 (row major)
// Block tile 128x128, 8 waves (4 M x 2 N), wave tile 32x64 -> 8 WMMAs / K-step.
// A tile is staged to LDS by the Tensor Data Mover (tensor_load_to_lds, wave 0,
// tracked with TENSORcnt); B tile staged transposed with regular loads.
// M % 128 == 0, N % 128 == 0, K % 32 == 0 (true for all shapes here).
// ---------------------------------------------------------------------------
__global__ __launch_bounds__(256) void k_gemm_bf16(const bf16* __restrict__ A,
                                                   const bf16* __restrict__ B,
                                                   float* __restrict__ C,
                                                   int M, int N, int K) {
  __shared__ __align__(16) bf16 sA[128][32];
  __shared__ __align__(16) bf16 sBt[128][32];  // B tile stored transposed [n][k]
  const int tid  = threadIdx.x;
  const int wave = tid >> 5;
  const int lane = tid & 31;
  const int lhi  = lane >> 4;   // which 16-lane half
  const int l16  = lane & 15;
  const int m0 = blockIdx.y * 128;
  const int n0 = blockIdx.x * 128;
  const int wm = (wave & 3) * 32;
  const int wn = (wave >> 2) * 64;

  v8f acc[2][4];
#pragma unroll
  for (int i = 0; i < 2; ++i)
#pragma unroll
    for (int j = 0; j < 4; ++j)
#pragma unroll
      for (int e = 0; e < 8; ++e) acc[i][j][e] = 0.0f;

  const int bk = tid & 31;           // B stage: k row 0..31
  const int bn = (tid >> 5) * 16;    //          n chunk 0..112
  const unsigned ldsA = lds_off(&sA[0][0]);

  for (int k0 = 0; k0 < K; k0 += 32) {
    // --- A tile via Tensor Data Mover: 2D tile 32 (K) x 128 (M) of 2B elems ---
    if (tid < 32) {  // wave 0 issues the TDM descriptor (EXEC ignored by TDM)
      const unsigned long long ga =
          (unsigned long long)(A + (size_t)m0 * K + k0);
      u32x4 g0;
      g0[0] = 1u;                                   // count=1, user mode
      g0[1] = ldsA;                                 // lds_addr (bytes)
      g0[2] = (unsigned)ga;                         // global_addr[31:0]
      g0[3] = (unsigned)(ga >> 32) | (2u << 30);    // addr[56:32] | type=2
      u32x8 g1;
      g1[0] = 1u << 16;                             // data_size=2B, no mask
      g1[1] = ((unsigned)K & 0xFFFFu) << 16;        // tensor_dim0[15:0]
      g1[2] = ((unsigned)K >> 16) |                 // tensor_dim0[31:16]
              (((unsigned)M & 0xFFFFu) << 16);      // tensor_dim1[15:0]
      g1[3] = ((unsigned)M >> 16) | (32u << 16);    // tensor_dim1 hi | tile_dim0=32
      g1[4] = 128u;                                 // tile_dim1=128, tile_dim2=0
      g1[5] = (unsigned)K;                          // tensor_dim0_stride[31:0]
      g1[6] = 0u;                                   // stride hi, dim1_stride lo
      g1[7] = 0u;
      asm volatile("tensor_load_to_lds %0, %1" :: "s"(g0), "s"(g1) : "memory");
    }

    // --- B tile: regular loads, transposed scatter into LDS ---
    const bf16* gB = B + (size_t)(k0 + bk) * N + n0 + bn;
    v8bf b0 = *(const v8bf*)gB;
    v8bf b1 = *(const v8bf*)(gB + 8);
#pragma unroll
    for (int j = 0; j < 8; ++j) {
      sBt[bn + j][bk]     = b0[j];
      sBt[bn + 8 + j][bk] = b1[j];
    }
    if (k0 + 32 < K)
      __builtin_prefetch(B + (size_t)(k0 + 32 + bk) * N + n0 + bn, 0, 1);

    if (tid < 32) __builtin_amdgcn_s_wait_tensorcnt(0);  // TDM done (wave 0)
    __syncthreads();

    v16bf af[2], bfr[4];
#pragma unroll
    for (int i = 0; i < 2; ++i) af[i]  = ld_frag(&sA[wm + i * 16 + l16][lhi * 8]);
#pragma unroll
    for (int j = 0; j < 4; ++j) bfr[j] = ld_frag(&sBt[wn + j * 16 + l16][lhi * 8]);
#pragma unroll
    for (int i = 0; i < 2; ++i)
#pragma unroll
      for (int j = 0; j < 4; ++j)
        acc[i][j] = __builtin_amdgcn_wmma_f32_16x16x32_bf16(
            false, af[i], false, bfr[j], (short)0, acc[i][j], false, false);
    __syncthreads();
  }

  // C/D layout: lanes 0-15 -> M=r, lanes 16-31 -> M=r+8; N = lane&15
#pragma unroll
  for (int i = 0; i < 2; ++i)
#pragma unroll
    for (int j = 0; j < 4; ++j) {
      const int rb = m0 + wm + i * 16 + lhi * 8;
      const int cc = n0 + wn + j * 16 + l16;
#pragma unroll
      for (int r = 0; r < 8; ++r)
        C[(size_t)(rb + r) * N + cc] = acc[i][j][r];
    }
}

// ---------------------------------------------------------------------------
// RoPE + head reorder:  [B,T,D] f32 -> [B,H,T,DH] bf16 (q scaled by 1/sqrt(DH))
// ---------------------------------------------------------------------------
__global__ void k_rope(const float* __restrict__ qf, const float* __restrict__ kf,
                       const float* __restrict__ vf, bf16* __restrict__ qo,
                       bf16* __restrict__ ko, bf16* __restrict__ vo, int T,
                       float qscale) {
  const size_t total = (size_t)BATCH * NHEAD * T * (DHEAD / 2);
  for (size_t i = (size_t)blockIdx.x * blockDim.x + threadIdx.x; i < total;
       i += (size_t)gridDim.x * blockDim.x) {
    const int d = (int)(i % (DHEAD / 2));
    size_t r = i / (DHEAD / 2);
    const int t = (int)(r % T);
    r /= T;
    const int hh = (int)(r % NHEAD);
    const int bb = (int)(r / NHEAD);
    const float inv = __expf(-((float)(2 * d) / (float)DHEAD) * __logf(10000.0f));
    float sn, cs;
    __sincosf((float)t * inv, &sn, &cs);
    const size_t src = ((size_t)bb * T + t) * DMODEL + hh * DHEAD;
    const size_t dst = (((size_t)bb * NHEAD + hh) * T + t) * DHEAD;
    const float q1 = qf[src + d], q2 = qf[src + d + 64];
    const float k1 = kf[src + d], k2 = kf[src + d + 64];
    qo[dst + d]      = (bf16)((q1 * cs - q2 * sn) * qscale);
    qo[dst + d + 64] = (bf16)((q2 * cs + q1 * sn) * qscale);
    ko[dst + d]      = (bf16)(k1 * cs - k2 * sn);
    ko[dst + d + 64] = (bf16)(k2 * cs + k1 * sn);
    vo[dst + d]      = (bf16)vf[src + d];
    vo[dst + d + 64] = (bf16)vf[src + d + 64];
  }
}

// ---------------------------------------------------------------------------
// Causal flash attention (bf16 WMMA, online softmax).
// grid = (T/64, H, B), block = 128 (4 waves); wave owns 16 q rows.
// K block staged with GLOBAL_LOAD_ASYNC_TO_LDS_B128 (ASYNCcnt).
// Q,K,V: [B,H,T,DH] bf16 (q pre-scaled).  Out: [B,T,D] bf16.
// ---------------------------------------------------------------------------
__global__ __launch_bounds__(128) void k_flash(const bf16* __restrict__ Q,
                                               const bf16* __restrict__ Kmat,
                                               const bf16* __restrict__ Vmat,
                                               bf16* __restrict__ O, int T) {
  __shared__ __align__(16) bf16 sK[32][DHEAD];    // key block, row major
  __shared__ __align__(16) bf16 sVt[DHEAD][32];   // V block transposed [dh][key]
  __shared__ float sP[4][16][32];                 // per-wave P scratch

  const int bb = blockIdx.z, hh = blockIdx.y;
  const int q0 = blockIdx.x * 64;
  const int tid = threadIdx.x;
  const int wave = tid >> 5, lane = tid & 31;
  const int lhi = lane >> 4, l16 = lane & 15;
  const int qw = q0 + wave * 16;

  const bf16* Qb = Q    + ((size_t)bb * NHEAD + hh) * T * DHEAD;
  const bf16* Kb = Kmat + ((size_t)bb * NHEAD + hh) * T * DHEAD;
  const bf16* Vb = Vmat + ((size_t)bb * NHEAD + hh) * T * DHEAD;

  // Q fragments for this wave's 16 rows (kept in VGPRs for the whole kernel)
  v16bf qfr[4];
#pragma unroll
  for (int d = 0; d < 4; ++d)
    qfr[d] = ld_frag(Qb + (size_t)(qw + l16) * DHEAD + d * 32 + lhi * 8);

  v8f acc[8];
#pragma unroll
  for (int j = 0; j < 8; ++j)
#pragma unroll
    for (int e = 0; e < 8; ++e) acc[j][e] = 0.0f;
  float mrow[8], lrow[8];
#pragma unroll
  for (int r = 0; r < 8; ++r) { mrow[r] = -1e30f; lrow[r] = 0.0f; }

  int kend = q0 + 64; if (kend > T) kend = T;
  for (int kb = 0; kb < kend; kb += 32) {
    // stage K block (32 x 128) with async LDS loads: 16B per lane per issue
    {
      const int kk = tid >> 2, cb = (tid & 3) * 32;
      const bf16* gp = Kb + (size_t)(kb + kk) * DHEAD + cb;
      const unsigned lo = lds_off(&sK[kk][cb]);
#pragma unroll
      for (int u = 0; u < 4; ++u) {
        const unsigned long long ga = (unsigned long long)(gp + u * 8);
        asm volatile("global_load_async_to_lds_b128 %0, %1, off"
                     :: "v"(lo + u * 16), "v"(ga) : "memory");
      }
    }
    // stage V block transposed (128 x 32) with regular loads
    {
      const int kk = tid & 31, nb = (tid >> 5) * 32;
      const bf16* gp = Vb + (size_t)(kb + kk) * DHEAD + nb;
#pragma unroll
      for (int u = 0; u < 4; ++u) {
        v8bf vv = *(const v8bf*)(gp + u * 8);
#pragma unroll
        for (int e = 0; e < 8; ++e) sVt[nb + u * 8 + e][kk] = vv[e];
      }
    }
    asm volatile("s_wait_asynccnt 0x0" ::: "memory");
    __syncthreads();

    // S = Q @ K^T for this 16x32 block (two 16x16 C tiles)
    v8f s0, s1;
#pragma unroll
    for (int e = 0; e < 8; ++e) { s0[e] = 0.0f; s1[e] = 0.0f; }
#pragma unroll
    for (int d = 0; d < 4; ++d) {
      s0 = __builtin_amdgcn_wmma_f32_16x16x32_bf16(
          false, qfr[d], false, ld_frag(&sK[l16][d * 32 + lhi * 8]),
          (short)0, s0, false, false);
      s1 = __builtin_amdgcn_wmma_f32_16x16x32_bf16(
          false, qfr[d], false, ld_frag(&sK[16 + l16][d * 32 + lhi * 8]),
          (short)0, s1, false, false);
    }

    // causal mask + online softmax (row lives in 16-lane half; shfl_xor 1/2/4/8)
#pragma unroll
    for (int r = 0; r < 8; ++r) {
      const int qi = qw + r + 8 * lhi;
      const float a0 = ((kb + l16)      <= qi) ? s0[r] : -1e30f;
      const float a1 = ((kb + 16 + l16) <= qi) ? s1[r] : -1e30f;
      float rm = fmaxf(a0, a1);
      rm = fmaxf(rm, __shfl_xor(rm, 1));
      rm = fmaxf(rm, __shfl_xor(rm, 2));
      rm = fmaxf(rm, __shfl_xor(rm, 4));
      rm = fmaxf(rm, __shfl_xor(rm, 8));
      const float mnew = fmaxf(mrow[r], rm);
      const float sc = __expf(mrow[r] - mnew);
      const float p0 = __expf(a0 - mnew);
      const float p1 = __expf(a1 - mnew);
      float rs = p0 + p1;
      rs += __shfl_xor(rs, 1);
      rs += __shfl_xor(rs, 2);
      rs += __shfl_xor(rs, 4);
      rs += __shfl_xor(rs, 8);
      lrow[r] = lrow[r] * sc + rs;
      mrow[r] = mnew;
#pragma unroll
      for (int j = 0; j < 8; ++j) acc[j][r] *= sc;
      sP[wave][r + 8 * lhi][l16]      = p0;
      sP[wave][r + 8 * lhi][16 + l16] = p1;
    }
    __syncthreads();

    // reshape P: C-layout (via LDS) -> A fragment 16x32 bf16
    v16bf pfr;
    {
      const float* pr = &sP[wave][l16][lhi * 8];
#pragma unroll
      for (int i = 0; i < 8; ++i) {
        pfr[i]     = (bf16)pr[i];
        pfr[i + 8] = (bf16)pr[i + 16];
      }
    }
    // O += P @ V
#pragma unroll
    for (int j = 0; j < 8; ++j)
      acc[j] = __builtin_amdgcn_wmma_f32_16x16x32_bf16(
          false, pfr, false, ld_frag(&sVt[j * 16 + l16][lhi * 8]),
          (short)0, acc[j], false, false);
    __syncthreads();
  }

  // normalize and write [B,T,D] bf16
#pragma unroll
  for (int r = 0; r < 8; ++r) {
    const int qi = qw + r + 8 * lhi;
    const float inv = 1.0f / lrow[r];
#pragma unroll
    for (int j = 0; j < 8; ++j)
      O[((size_t)bb * T + qi) * DMODEL + hh * DHEAD + j * 16 + l16] =
          (bf16)(acc[j][r] * inv);
  }
}

// ---------------------------------------------------------------------------
// SwiGLU combine: out_bf = bf16(silu(g) * u)
// ---------------------------------------------------------------------------
__global__ void k_swiglu(const float* __restrict__ g, const float* __restrict__ u,
                         bf16* __restrict__ o, size_t n) {
  for (size_t i = (size_t)blockIdx.x * blockDim.x + threadIdx.x; i < n;
       i += (size_t)gridDim.x * blockDim.x) {
    const float gv = g[i];
    o[i] = (bf16)((gv / (1.0f + __expf(-gv))) * u[i]);
  }
}

__global__ void k_add(const float* __restrict__ a, const float* __restrict__ b,
                      float* __restrict__ o, size_t n) {
  for (size_t i = (size_t)blockIdx.x * blockDim.x + threadIdx.x; i < n;
       i += (size_t)gridDim.x * blockDim.x)
    o[i] = a[i] + b[i];
}

// ---------------------------------------------------------------------------
// Surprise router: g = sigmoid(mean((ar-pr)^2) - mean(ar^2)), ar = proc - x
// ---------------------------------------------------------------------------
__global__ __launch_bounds__(256) void k_router(const float* __restrict__ x,
                                                const float* __restrict__ proc,
                                                const float* __restrict__ prior,
                                                float* __restrict__ g, int Dn) {
  __shared__ float r1[256], r2[256];
  const int row = blockIdx.x;
  const size_t base = (size_t)row * Dn;
  float s1 = 0.0f, s2 = 0.0f;
  for (int c = threadIdx.x; c < Dn; c += 256) {
    const float ar = proc[base + c] - x[base + c];
    const float df = ar - prior[base + c];
    s1 += ar * ar;
    s2 += df * df;
  }
  r1[threadIdx.x] = s1; r2[threadIdx.x] = s2; __syncthreads();
  for (int k = 128; k > 0; k >>= 1) {
    if (threadIdx.x < k) {
      r1[threadIdx.x] += r1[threadIdx.x + k];
      r2[threadIdx.x] += r2[threadIdx.x + k];
    }
    __syncthreads();
  }
  if (threadIdx.x == 0) {
    const float Dst = r1[0] / (float)Dn, Dch = r2[0] / (float)Dn;
    g[row] = 1.0f / (1.0f + __expf(-(Dch - Dst)));
  }
}

// deterministic two-stage loss reduction: mean((x + prior - proc)^2)
__global__ __launch_bounds__(256) void k_loss_part(const float* __restrict__ x,
                                                   const float* __restrict__ prior,
                                                   const float* __restrict__ proc,
                                                   float* __restrict__ part, size_t n) {
  __shared__ float red[256];
  float s = 0.0f;
  for (size_t i = (size_t)blockIdx.x * blockDim.x + threadIdx.x; i < n;
       i += (size_t)gridDim.x * blockDim.x) {
    const float d = x[i] + prior[i] - proc[i];
    s += d * d;
  }
  red[threadIdx.x] = s; __syncthreads();
  for (int k = 128; k > 0; k >>= 1) {
    if (threadIdx.x < k) red[threadIdx.x] += red[threadIdx.x + k];
    __syncthreads();
  }
  if (threadIdx.x == 0) part[blockIdx.x] = red[0];
}

__global__ __launch_bounds__(256) void k_loss_final(const float* __restrict__ part,
                                                    float* __restrict__ loss, int nb,
                                                    float invn) {
  __shared__ float red[256];
  float s = 0.0f;
  for (int i = threadIdx.x; i < nb; i += 256) s += part[i];
  red[threadIdx.x] = s; __syncthreads();
  for (int k = 128; k > 0; k >>= 1) {
    if (threadIdx.x < k) red[threadIdx.x] += red[threadIdx.x + k];
    __syncthreads();
  }
  if (threadIdx.x == 0) loss[0] = red[0] * invn;
}

// ---------------------------------------------------------------------------
// Per-batch top-k via in-LDS bitonic sort (descending), T = 2048, 1024 thr.
// ---------------------------------------------------------------------------
__global__ __launch_bounds__(1024) void k_topk(const float* __restrict__ g,
                                               int* __restrict__ idx,
                                               float* __restrict__ gate, int T,
                                               int Ksel) {
  __shared__ float sv[SEQ];
  __shared__ int   si[SEQ];
  const int bb = blockIdx.x;
  const int tid = threadIdx.x;
  for (int i = tid; i < T; i += 1024) { sv[i] = g[(size_t)bb * T + i]; si[i] = i; }
  __syncthreads();
  for (int k = 2; k <= T; k <<= 1) {
    for (int j = k >> 1; j > 0; j >>= 1) {
      for (int i = tid; i < T; i += 1024) {
        const int ixj = i ^ j;
        if (ixj > i) {
          const bool up = ((i & k) == 0);  // descending in "up" segments
          const float a = sv[i], c = sv[ixj];
          if (up ? (a < c) : (a > c)) {
            sv[i] = c; sv[ixj] = a;
            const int t = si[i]; si[i] = si[ixj]; si[ixj] = t;
          }
        }
      }
      __syncthreads();
    }
  }
  for (int i = tid; i < Ksel; i += 1024) {
    idx[bb * Ksel + i]  = si[i];
    gate[bb * Ksel + i] = sv[i];
  }
}

__global__ void k_gather(const float* __restrict__ proc, const int* __restrict__ idx,
                         float* __restrict__ sel) {
  const size_t total = (size_t)BATCH * KSEL * DMODEL;
  for (size_t i = (size_t)blockIdx.x * blockDim.x + threadIdx.x; i < total;
       i += (size_t)gridDim.x * blockDim.x) {
    const int d = (int)(i % DMODEL);
    const size_t r = i / DMODEL;
    const int j = (int)(r % KSEL);
    const int bb = (int)(r / KSEL);
    sel[i] = proc[((size_t)bb * SEQ + idx[bb * KSEL + j]) * DMODEL + d];
  }
}

__global__ void k_scatter(float* __restrict__ out, const float* __restrict__ sel,
                          const float* __restrict__ selo,
                          const float* __restrict__ gate, const int* __restrict__ idx) {
  const size_t total = (size_t)BATCH * KSEL * DMODEL;
  for (size_t i = (size_t)blockIdx.x * blockDim.x + threadIdx.x; i < total;
       i += (size_t)gridDim.x * blockDim.x) {
    const int d = (int)(i % DMODEL);
    const size_t r = i / DMODEL;
    const int j = (int)(r % KSEL);
    const int bb = (int)(r / KSEL);
    const float s = sel[i];
    const float upd = s + gate[bb * KSEL + j] * (selo[i] - s);
    out[((size_t)bb * SEQ + idx[bb * KSEL + j]) * DMODEL + d] = upd;
  }
}

// ---------------------------------------------------------------------------
// Host orchestration
// ---------------------------------------------------------------------------
static void launch_gemm(const bf16* A, const bf16* B, float* C, int M, int N, int K,
                        hipStream_t s) {
  dim3 g(N / 128, M / 128);
  k_gemm_bf16<<<g, 256, 0, s>>>(A, B, C, M, N, K);
}

struct Wts { const float *ln1, *ln2; const bf16 *wq, *wk, *wv, *wo, *wg, *wu, *wd; };
struct Scr {
  bf16 *xn, *qb, *kb, *vb, *ab, *fb;
  float *qf, *kf, *vf, *ao, *h, *g, *u, *fo;
};

static void run_decoder(const float* x, float* out, int Tn, const Wts& W,
                        const Scr& S, hipStream_t st) {
  const int M = BATCH * Tn;
  k_rmsnorm<<<M, 256, 0, st>>>(x, W.ln1, S.xn, DMODEL);
  launch_gemm(S.xn, W.wq, S.qf, M, DMODEL, DMODEL, st);
  launch_gemm(S.xn, W.wk, S.kf, M, DMODEL, DMODEL, st);
  launch_gemm(S.xn, W.wv, S.vf, M, DMODEL, DMODEL, st);
  k_rope<<<1024, 256, 0, st>>>(S.qf, S.kf, S.vf, S.qb, S.kb, S.vb, Tn,
                               0.08838834764831845f /* 1/sqrt(128) */);
  k_flash<<<dim3(Tn / 64, NHEAD, BATCH), 128, 0, st>>>(S.qb, S.kb, S.vb, S.ab, Tn);
  launch_gemm(S.ab, W.wo, S.ao, M, DMODEL, DMODEL, st);
  k_resid_rms<<<M, 256, 0, st>>>(x, S.ao, W.ln2, S.h, S.xn, DMODEL);
  launch_gemm(S.xn, W.wg, S.g, M, FFD, DMODEL, st);
  launch_gemm(S.xn, W.wu, S.u, M, FFD, DMODEL, st);
  k_swiglu<<<2048, 256, 0, st>>>(S.g, S.u, S.fb, (size_t)M * FFD);
  launch_gemm(S.fb, W.wd, S.fo, M, DMODEL, FFD, st);
  k_add<<<1024, 256, 0, st>>>(S.h, S.fo, out, (size_t)M * DMODEL);
}

extern "C" void kernel_launch(void* const* d_in, const int* in_sizes, int n_in,
                              void* d_out, int out_size, void* d_ws, size_t ws_size,
                              hipStream_t stream) {
  (void)in_sizes; (void)n_in; (void)out_size; (void)ws_size;
  const float* f[23];
  for (int i = 0; i < 23; ++i) f[i] = (const float*)d_in[i];
  const float* x0 = f[0];

  char* p = (char*)d_ws;
  auto alloc = [&](size_t bytes) -> char* {
    char* r = p;
    p += (bytes + 255) & ~(size_t)255;
    return r;
  };

  const size_t DD = (size_t)DMODEL * DMODEL;
  const size_t DF = (size_t)DMODEL * FFD;
  const size_t DP = (size_t)DMODEL * PFFD;
  const int M1 = BATCH * SEQ;
  const int M2 = BATCH * KSEL;

  // bf16 weight copies: d1 wq..wd (0-6), d2 wq..wd (7-13), p_wg/p_wu/p_wd (14-16)
  const float* wsrc[17] = { f[3],  f[4],  f[5],  f[6],  f[7],  f[8],  f[9],
                            f[12], f[13], f[14], f[15], f[16], f[17], f[18],
                            f[20], f[21], f[22] };
  const size_t wn[17] = { DD, DD, DD, DD, DF, DF, DF,
                          DD, DD, DD, DD, DF, DF, DF,
                          DP, DP, DP };
  bf16* wb[17];
  for (int i = 0; i < 17; ++i) {
    wb[i] = (bf16*)alloc(wn[i] * sizeof(bf16));
    k_cvt<<<1024, 256, 0, stream>>>(wsrc[i], wb[i], wn[i]);
  }

  Scr S;
  S.xn = (bf16*)alloc((size_t)M1 * DMODEL * sizeof(bf16));
  S.qb = (bf16*)alloc((size_t)M1 * DMODEL * sizeof(bf16));
  S.kb = (bf16*)alloc((size_t)M1 * DMODEL * sizeof(bf16));
  S.vb = (bf16*)alloc((size_t)M1 * DMODEL * sizeof(bf16));
  S.ab = (bf16*)alloc((size_t)M1 * DMODEL * sizeof(bf16));
  S.fb = (bf16*)alloc((size_t)M1 * FFD * sizeof(bf16));
  S.qf = (float*)alloc((size_t)M1 * DMODEL * sizeof(float));
  S.kf = (float*)alloc((size_t)M1 * DMODEL * sizeof(float));
  S.vf = (float*)alloc((size_t)M1 * DMODEL * sizeof(float));
  S.ao = (float*)alloc((size_t)M1 * DMODEL * sizeof(float));
  S.h  = (float*)alloc((size_t)M1 * DMODEL * sizeof(float));
  S.g  = (float*)alloc((size_t)M1 * FFD * sizeof(float));
  S.u  = (float*)alloc((size_t)M1 * FFD * sizeof(float));
  S.fo = (float*)alloc((size_t)M1 * DMODEL * sizeof(float));
  float* prior_out = (float*)alloc((size_t)M1 * DMODEL * sizeof(float));
  float* sel  = (float*)alloc((size_t)M2 * DMODEL * sizeof(float));
  float* selo = (float*)alloc((size_t)M2 * DMODEL * sizeof(float));
  int*   idx  = (int*)alloc((size_t)BATCH * KSEL * sizeof(int));
  float* gate = (float*)alloc((size_t)BATCH * KSEL * sizeof(float));
  float* part = (float*)alloc(1024 * sizeof(float));

  float* outF  = (float*)d_out;
  float* procF = outF;                               // processed -> final (in place)
  float* lossF = outF + (size_t)M1 * DMODEL;         // scalar prior_loss
  float* gcont = lossF + 1;                          // [B,T] gate values

  const Wts W1 = { f[1],  f[2],  wb[0], wb[1], wb[2], wb[3], wb[4],  wb[5],  wb[6] };
  const Wts W2 = { f[10], f[11], wb[7], wb[8], wb[9], wb[10], wb[11], wb[12], wb[13] };

  // decision layer (writes processed straight into d_out)
  run_decoder(x0, procF, SEQ, W1, S, stream);

  // prior network (reuses FFN scratch; PFFD <= FFD)
  k_rmsnorm<<<M1, 256, 0, stream>>>(x0, f[19], S.xn, DMODEL);
  launch_gemm(S.xn, wb[14], S.g, M1, PFFD, DMODEL, stream);
  launch_gemm(S.xn, wb[15], S.u, M1, PFFD, DMODEL, stream);
  k_swiglu<<<1024, 256, 0, stream>>>(S.g, S.u, S.fb, (size_t)M1 * PFFD);
  launch_gemm(S.fb, wb[16], prior_out, M1, DMODEL, PFFD, stream);

  // router gate + prior loss
  k_router<<<M1, 256, 0, stream>>>(x0, procF, prior_out, gcont, DMODEL);
  k_loss_part<<<1024, 256, 0, stream>>>(x0, prior_out, procF, part,
                                        (size_t)M1 * DMODEL);
  k_loss_final<<<1, 256, 0, stream>>>(part, lossF, 1024,
                                      1.0f / ((float)M1 * (float)DMODEL));

  // top-k selection, gather, dynamic layer, soft-gated scatter
  k_topk<<<BATCH, 1024, 0, stream>>>(gcont, idx, gate, SEQ, KSEL);
  k_gather<<<1024, 256, 0, stream>>>(procF, idx, sel);
  run_decoder(sel, selo, KSEL, W2, S, stream);
  k_scatter<<<1024, 256, 0, stream>>>(procF, sel, selo, gate, idx);
}
